// GraphSage_56075093016832
// MI455X (gfx1250) — compile-verified, matrix operand order checked by reading, not run on hardware
//
#include <hip/hip_runtime.h>
#include <stdint.h>

// ---------------- types for WMMA ----------------
typedef __bf16 v16bf __attribute__((ext_vector_type(16)));
typedef float  v8f   __attribute__((ext_vector_type(8)));

struct __attribute__((aligned(16))) U4 { unsigned x, y, z, w; };
struct U32x8 { U4 a, b; };

// ---------------- problem sizes ----------------
#define NB   4096          // batch
#define KF   10            // neighbor fanout
#define INS  256
#define OUTS 256
#define KD   512           // 2*256 (concat [self, agg])
#define M1   (NB + NB*KF)  // 45056 layer-1 rows

// ---------------- bf16 split helpers ----------------
__device__ __forceinline__ unsigned short f32_to_bf16_rne(float x) {
  unsigned u = __builtin_bit_cast(unsigned, x);
  unsigned r = 0x7FFFu + ((u >> 16) & 1u);
  u += r;
  return (unsigned short)(u >> 16);
}
__device__ __forceinline__ float bf16_to_f32(unsigned short h) {
  unsigned u = ((unsigned)h) << 16;
  return __builtin_bit_cast(float, u);
}
__device__ __forceinline__ void split_store(unsigned short* ph, unsigned short* pl, float x) {
  unsigned short h = f32_to_bf16_rne(x);
  float rem = x - bf16_to_f32(h);
  *ph = h;
  *pl = f32_to_bf16_rne(rem);
}

// Load one 16-bit WMMA A/B fragment for K-tile kb.
// ISA layout (16-bit A 16x32): lane half 0 -> elems 0..7 = K[kb..kb+7], elems 8..15 = K[kb+16..kb+23]
//                              lane half 1 -> same +8. Two contiguous 16B segments per lane.
__device__ __forceinline__ v16bf load_frag(const unsigned short* __restrict__ row, int kb, int half) {
  U32x8 u;
  u.a = *(const U4*)(row + kb + half * 8);
  u.b = *(const U4*)(row + kb + 16 + half * 8);
  return __builtin_bit_cast(v16bf, u);
}

// ---------------- kernel 1: split weights to bf16 hi/lo ----------------
__global__ void sage_split_w(const float* __restrict__ W,
                             unsigned short* __restrict__ Whi,
                             unsigned short* __restrict__ Wlo, int n) {
  int i = blockIdx.x * blockDim.x + threadIdx.x;
  if (i < n) split_store(Whi + i, Wlo + i, W[i]);
}

// ---------------- kernel 2: build layer-1 input rows (gather + mean), split bf16 ----
// Row r < NB:        self = emb[node_batch[r]],       agg = mean_k emb[nb_l1_self[r][k]]
// Row r >= NB (t=r-NB): self = emb[nb_l2.flat[t]],    agg = mean_k emb[nb_l1_neigh.flat[t][k]]
__global__ void sage_build_x1(const float* __restrict__ emb,
                              const int* __restrict__ node_batch,
                              const int* __restrict__ nb_l1_self,
                              const int* __restrict__ nb_l2,
                              const int* __restrict__ nb_l1_neigh,
                              unsigned short* __restrict__ X1hi,
                              unsigned short* __restrict__ X1lo) {
  int r = blockIdx.x;          // 0 .. M1-1
  int c = threadIdx.x;         // 0 .. 255
  int self_idx;
  const int* neigh;
  if (r < NB) {
    self_idx = node_batch[r];
    neigh    = nb_l1_self + (size_t)r * KF;
  } else {
    int t = r - NB;            // = b*KF + j
    self_idx = nb_l2[t];
    neigh    = nb_l1_neigh + (size_t)t * KF;
  }
  float self = emb[(size_t)self_idx * INS + c];
  float s = 0.f;
#pragma unroll
  for (int k = 0; k < KF; ++k) s += emb[(size_t)neigh[k] * INS + c];
  s *= (1.0f / KF);
  size_t base = (size_t)r * KD;
  split_store(X1hi + base + c,       X1lo + base + c,       self);
  split_store(X1hi + base + INS + c, X1lo + base + INS + c, s);
}

// ---------------- kernel 3: C[M x 256] = relu(A[M x 512] @ W[256 x 512]^T) ----------
// A, W pre-split into bf16 hi/lo. Block = 128 threads = 4 waves; block owns a 16-row
// M tile, wave w owns N columns [64w, 64w+64) as four 16x16 WMMA tiles.
__global__ void sage_gemm_relu(const unsigned short* __restrict__ Ahi,
                               const unsigned short* __restrict__ Alo,
                               const unsigned short* __restrict__ Whi,
                               const unsigned short* __restrict__ Wlo,
                               float* __restrict__ C) {
  const int N = OUTS;
  int m0   = blockIdx.x * 16;
  int wave = threadIdx.x >> 5;
  int lane = threadIdx.x & 31;
  int half = lane >> 4;
  int l15  = lane & 15;

  const unsigned short* arow_hi = Ahi + (size_t)(m0 + l15) * KD;
  const unsigned short* arow_lo = Alo + (size_t)(m0 + l15) * KD;
  int n0w = wave * 64;

  v8f acc0 = {}, acc1 = {}, acc2 = {}, acc3 = {};
  v8f* acc[4] = { &acc0, &acc1, &acc2, &acc3 };

  for (int kb = 0; kb < KD; kb += 32) {
    v16bf ahi = load_frag(arow_hi, kb, half);
    v16bf alo = load_frag(arow_lo, kb, half);
#pragma unroll
    for (int nt = 0; nt < 4; ++nt) {
      int n = n0w + nt * 16 + l15;   // output column owned by this lane
      const unsigned short* wrow_hi = Whi + (size_t)n * KD;
      const unsigned short* wrow_lo = Wlo + (size_t)n * KD;
      v16bf bhi = load_frag(wrow_hi, kb, half);
      v16bf blo = load_frag(wrow_lo, kb, half);
      v8f a = *acc[nt];
      a = __builtin_amdgcn_wmma_f32_16x16x32_bf16(false, ahi, false, bhi, (short)0, a, false, false);
      a = __builtin_amdgcn_wmma_f32_16x16x32_bf16(false, ahi, false, blo, (short)0, a, false, false);
      a = __builtin_amdgcn_wmma_f32_16x16x32_bf16(false, alo, false, bhi, (short)0, a, false, false);
      *acc[nt] = a;
    }
  }

  // f32 C/D layout: lane&15 = N, VGPR r -> M = r + 8*(lane>=16)
#pragma unroll
  for (int nt = 0; nt < 4; ++nt) {
    int n = n0w + nt * 16 + l15;
    v8f a = *acc[nt];
#pragma unroll
    for (int r = 0; r < 8; ++r) {
      int m = m0 + r + 8 * half;
      float v = a[r];
      C[(size_t)m * N + n] = v > 0.f ? v : 0.f;
    }
  }
}

// ---------------- kernel 4: build layer-2 input rows from H1, split bf16 ----------
__global__ void sage_build_x2(const float* __restrict__ H1,
                              unsigned short* __restrict__ X2hi,
                              unsigned short* __restrict__ X2lo) {
  int b = blockIdx.x;          // 0 .. NB-1
  int c = threadIdx.x;         // 0 .. 255
  float self = H1[(size_t)b * OUTS + c];
  float s = 0.f;
#pragma unroll
  for (int j = 0; j < KF; ++j)
    s += H1[(size_t)(NB + b * KF + j) * OUTS + c];
  s *= (1.0f / KF);
  size_t base = (size_t)b * KD;
  split_store(X2hi + base + c,        X2lo + base + c,        self);
  split_store(X2hi + base + OUTS + c, X2lo + base + OUTS + c, s);
}

// ---------------- launcher ----------------
extern "C" void kernel_launch(void* const* d_in, const int* in_sizes, int n_in,
                              void* d_out, int out_size, void* d_ws, size_t ws_size,
                              hipStream_t stream) {
  const float* emb        = (const float*)d_in[0];
  const float* W1         = (const float*)d_in[1];
  const float* W2         = (const float*)d_in[2];
  const int* node_batch   = (const int*)d_in[3];
  const int* nb_l1_self   = (const int*)d_in[4];
  const int* nb_l2        = (const int*)d_in[5];
  const int* nb_l1_neigh  = (const int*)d_in[6];
  float* out = (float*)d_out;
  char* ws = (char*)d_ws;

  // workspace layout (bytes)
  const size_t X1_BYTES = (size_t)M1 * KD * 2;        // 46,137,344 (bf16)
  unsigned short* X1hi = (unsigned short*)(ws);
  unsigned short* X1lo = (unsigned short*)(ws + X1_BYTES);
  float*          H1   = (float*)(ws + 2 * X1_BYTES); // 46,137,344 (f32)
  // X2 overlays the (dead after GEMM1) X1 region
  unsigned short* X2hi = (unsigned short*)(ws);
  unsigned short* X2lo = (unsigned short*)(ws + (size_t)NB * KD * 2);
  unsigned short* W1hi = (unsigned short*)(ws + 3 * X1_BYTES);
  unsigned short* W1lo = W1hi + (size_t)OUTS * KD;
  unsigned short* W2hi = W1lo + (size_t)OUTS * KD;
  unsigned short* W2lo = W2hi + (size_t)OUTS * KD;

  const int WN = OUTS * KD;  // 131072 weight elements
  sage_split_w<<<(WN + 255) / 256, 256, 0, stream>>>(W1, W1hi, W1lo, WN);
  sage_split_w<<<(WN + 255) / 256, 256, 0, stream>>>(W2, W2hi, W2lo, WN);

  sage_build_x1<<<M1, 256, 0, stream>>>(emb, node_batch, nb_l1_self, nb_l2,
                                        nb_l1_neigh, X1hi, X1lo);

  sage_gemm_relu<<<M1 / 16, 128, 0, stream>>>(X1hi, X1lo, W1hi, W1lo, H1);

  sage_build_x2<<<NB, 256, 0, stream>>>(H1, X2hi, X2lo);

  sage_gemm_relu<<<NB / 16, 128, 0, stream>>>(X2hi, X2lo, W2hi, W2lo, out);
}
